// ContrastiveLoss_85177791414715
// MI455X (gfx1250) — matrix operand order repaired and checked by reading, not compiled, not run on hardware
//
#include <hip/hip_runtime.h>
#include <hip/hip_bf16.h>
#include <math.h>

typedef __attribute__((ext_vector_type(16))) _Float16 v16h;
typedef __attribute__((ext_vector_type(8)))  float    v8f;

#define N_ROWS   16384
#define DIM      128
#define NGRP     1024          // labels in [0,1000), padded to pow2
#define INV_T    10.0f         // 1 / TEMPERATURE
#define EPSN     1e-12f
#define EPSL     1e-8f

// workspace layout (in floats)
#define WS_S     0                         // NGRP*DIM label sums
#define WS_SSUM  (NGRP*DIM)                // DIM     global sum
#define WS_INV   (WS_SSUM + DIM)           // N_ROWS  1/||x||
#define WS_PART  (WS_INV + N_ROWS)         // N_ROWS/16 per-wave partial losses
#define WS_TOTAL (WS_PART + N_ROWS/16)

__global__ void k_zero(float* ws, int n) {
    int i = blockIdx.x * blockDim.x + threadIdx.x;
    if (i < n) ws[i] = 0.0f;
}

// one wave per row: L2-normalize, record inv-norm, scatter-add into label-sum table
__global__ void k_norm_scatter(const float* __restrict__ f,
                               const int* __restrict__ lab,
                               float* __restrict__ ws) {
    const int wave = (blockIdx.x * blockDim.x + threadIdx.x) >> 5;
    const int lane = threadIdx.x & 31;
    const float4 v = *(const float4*)(f + (size_t)wave * DIM + lane * 4);
    float ss = v.x * v.x + v.y * v.y + v.z * v.z + v.w * v.w;
    #pragma unroll
    for (int m = 16; m >= 1; m >>= 1) ss += __shfl_xor(ss, m, 32);
    const float inv = 1.0f / fmaxf(sqrtf(ss), EPSN);   // torch F.normalize semantics
    if (lane == 0) ws[WS_INV + wave] = inv;
    const int g = lab[wave] & (NGRP - 1);
    float* Sg = ws + WS_S + g * DIM + lane * 4;
    unsafeAtomicAdd(Sg + 0, v.x * inv);
    unsafeAtomicAdd(Sg + 1, v.y * inv);
    unsafeAtomicAdd(Sg + 2, v.z * inv);
    unsafeAtomicAdd(Sg + 3, v.w * inv);
}

// s[d] = sum_g S[g][d]
__global__ void k_reduce_s(float* __restrict__ ws) {
    const int d = threadIdx.x;   // 128 threads
    float acc = 0.0f;
    for (int g = 0; g < NGRP; ++g) acc += ws[WS_S + g * DIM + d];
    ws[WS_SSUM + d] = acc;
}

// one wave per 16-row tile: pos/tot dots via v_wmma_f32_16x16x32_f16 diagonals
__global__ void k_wmma_loss(const float* __restrict__ f,
                            const int* __restrict__ lab,
                            float* __restrict__ ws) {
    const int lane  = threadIdx.x & 31;
    const int waveG = blockIdx.x * (blockDim.x >> 5) + (threadIdx.x >> 5);
    const int tile  = waveG * 16;
    const int half  = lane >> 4;        // 0 or 1 (which half-wave)
    const int sub   = lane & 15;

    const int   rowA = tile + sub;                 // A: lane holds row M=sub
    const float invA = ws[WS_INV + rowA];
    const int   g    = lab[rowA] & (NGRP - 1);     // B: lane's column = row 'sub' of tile
    const float* Arow = f + (size_t)rowA * DIM;
    const float* Brow = ws + WS_S + g * DIM;
    const float* Srow = ws + WS_SSUM;

    v8f c1 = {};   // diag -> pos dots
    v8f c2 = {};   // diag -> tot dots
    #pragma unroll
    for (int kc = 0; kc < 4; ++kc) {
        const int kb = kc * 32;
        v16h a, b, bs;
        // A k order used here (contiguous, same as B): halves 0..7 = K[kb + half*16 + j],
        //                          halves 8..15 = K[kb + half*16 + 8 + j]
        #pragma unroll
        for (int j = 0; j < 8; ++j) {
            a[j]     = (_Float16)(Arow[kb + half * 16 + j]     * invA);
            a[8 + j] = (_Float16)(Arow[kb + half * 16 + 8 + j] * invA);
        }
        // B k order used here (contiguous, same as A): lane = column N=sub, 16 consecutive K per half-wave
        #pragma unroll
        for (int j = 0; j < 16; ++j) {
            b[j]  = (_Float16)(Brow[kb + half * 16 + j]);
            bs[j] = (_Float16)(Srow[kb + half * 16 + j]);
        }
        c1 = __builtin_amdgcn_wmma_f32_16x16x32_f16(false, a, false, b,  (short)0, c1, false, false);
        c2 = __builtin_amdgcn_wmma_f32_16x16x32_f16(false, a, false, bs, (short)0, c2, false, false);
    }

    // D layout: VGPR r, lanes 0-15 -> (M=r, N=lane); lanes 16-31 -> (M=r+8, N=lane-16)
    // Diagonal D[n][n]: n<8 -> lane n, reg n ; n>=8 -> lane n+16, reg n-8.
    // Source lanes are 0..7 and 24..31; on both, reg index = lane & 7.
    const int ridx = lane & 7;
    float d1 = c1[0], d2 = c2[0];
    #pragma unroll
    for (int j = 1; j < 8; ++j) {
        d1 = (ridx == j) ? c1[j] : d1;
        d2 = (ridx == j) ? c2[j] : d2;
    }
    const int src = (sub < 8) ? sub : (sub + 16);
    const float pos = __shfl(d1, src, 32) * INV_T;
    const float tot = __shfl(d2, src, 32) * INV_T;

    float loss = (half == 0) ? -logf(pos / (tot + EPSL)) : 0.0f;
    #pragma unroll
    for (int m = 16; m >= 1; m >>= 1) loss += __shfl_xor(loss, m, 32);
    if (lane == 0) ws[WS_PART + waveG] = loss;   // unique slot per wave: deterministic
}

__global__ void k_final(const float* __restrict__ part, float* __restrict__ out) {
    __shared__ float sm[256];
    const int t = threadIdx.x;
    sm[t] = part[t] + part[t + 256] + part[t + 512] + part[t + 768];
    __syncthreads();
    for (int s = 128; s > 0; s >>= 1) {
        if (t < s) sm[t] += sm[t + s];
        __syncthreads();
    }
    if (t == 0) out[0] = sm[0] / (float)N_ROWS;
}

extern "C" void kernel_launch(void* const* d_in, const int* in_sizes, int n_in,
                              void* d_out, int out_size, void* d_ws, size_t ws_size,
                              hipStream_t stream) {
    const float* feat = (const float*)d_in[0];
    const int*   lab  = (const int*)d_in[1];
    float*       out  = (float*)d_out;
    float*       ws   = (float*)d_ws;

    k_zero<<<(WS_TOTAL + 255) / 256, 256, 0, stream>>>(ws, WS_TOTAL);
    k_norm_scatter<<<N_ROWS / 8, 256, 0, stream>>>(feat, lab, ws);   // 8 waves/block, 1 row/wave
    k_reduce_s<<<1, 128, 0, stream>>>(ws);
    k_wmma_loss<<<N_ROWS / 64, 128, 0, stream>>>(feat, lab, ws);     // 4 waves/block, 16 rows/wave
    k_final<<<1, 256, 0, stream>>>(ws + WS_PART, out);
}
